// MoeLayer_56246891709093
// MI455X (gfx1250) — matrix operand order repaired and checked by reading, not compiled, not run on hardware
//
#include <hip/hip_runtime.h>
#include <hip/hip_bf16.h>

// ---------------- problem constants ----------------
#define NTOK 8192
#define DDIM 1024
#define HDIM 4096
#define NEXP 8
// top-K = 2

// ---------------- types ----------------
typedef __bf16 v16bf __attribute__((ext_vector_type(16)));
typedef float  v8f   __attribute__((ext_vector_type(8)));
typedef unsigned u32x4 __attribute__((ext_vector_type(4)));
typedef int      i32x8 __attribute__((ext_vector_type(8)));
typedef int      i32x4 __attribute__((ext_vector_type(4)));

union Frag { v16bf v; uint4 q[2]; };

static __device__ __forceinline__ unsigned short to_bf(float f) {
    unsigned u = __builtin_bit_cast(unsigned, f);
    u += 0x7FFFu + ((u >> 16) & 1u);           // round-to-nearest-even
    return (unsigned short)(u >> 16);
}

static __device__ __forceinline__ float gelu_tanh(float x) {
    float x3 = x * x * x;
    return 0.5f * x * (1.0f + tanhf(0.7978845608028654f * (x + 0.044715f * x3)));
}

// ---------------- TDM: async 2D tile load (global -> LDS), bf16 ----------------
// tile: 32 elems (64B) per row, tile_rows rows. LDS gets +16B pad per 64B row
// (pad_interval=3 -> every 16 dwords, pad_amount=3 -> 4 dwords) => 80B row stride,
// matching LDA/LDB = 40 bf16 elems below.
static __device__ __forceinline__ void tdm_load_2d(unsigned lds_off, const void* gptr,
                                                   unsigned tensor_d0, unsigned tensor_d1,
                                                   unsigned stride_elems, unsigned tile_rows) {
    unsigned long long ga = (unsigned long long)(uintptr_t)gptr;
    u32x4 g0 = { 1u,                                   // count=1, user-mode load
                 lds_off,                              // LDS byte address
                 (unsigned)ga,                         // global addr [31:0]
                 (unsigned)((ga >> 32) & 0x01FFFFFFull) | 0x80000000u }; // [56:32] | type=2
    i32x8 g1;
    g1[0] = (1 << 16)      // data_size = 2 bytes
          | (1 << 20)      // pad_enable
          | (3 << 22)      // pad_interval: 16 dwords (64B)
          | (3 << 25);     // pad_amount:   4 dwords (16B)
    g1[1] = (int)(tensor_d0 << 16);                          // tensor_dim0[15:0]
    g1[2] = (int)((tensor_d0 >> 16) | (tensor_d1 << 16));    // dim0[31:16] | dim1[15:0]
    g1[3] = (int)((tensor_d1 >> 16) | (32u << 16));          // dim1[31:16] | tile_dim0=32
    g1[4] = (int)tile_rows;                                  // tile_dim1 (tile_dim2=0)
    g1[5] = (int)stride_elems;                               // tensor_dim0_stride[31:0]
    g1[6] = 0;
    g1[7] = 0;
    i32x4 gz4 = {0, 0, 0, 0};
    i32x8 gz8 = {0, 0, 0, 0, 0, 0, 0, 0};
    __builtin_amdgcn_tensor_load_to_lds(g0, g1, gz4, gz4, gz8, 0);
}

// ---------------- pre-pass: X fp32 -> bf16 ----------------
__global__ __launch_bounds__(256)
void convx_k(const float* __restrict__ src, unsigned short* __restrict__ dst) {
    size_t i = ((size_t)blockIdx.x * 256 + threadIdx.x) * 8;
    float4 f0 = *(const float4*)(src + i);
    float4 f1 = *(const float4*)(src + i + 4);
    union { unsigned short s[8]; uint4 q; } pk;
    pk.s[0] = to_bf(f0.x); pk.s[1] = to_bf(f0.y); pk.s[2] = to_bf(f0.z); pk.s[3] = to_bf(f0.w);
    pk.s[4] = to_bf(f1.x); pk.s[5] = to_bf(f1.y); pk.s[6] = to_bf(f1.z); pk.s[7] = to_bf(f1.w);
    *(uint4*)(dst + i) = pk.q;
}

// ---------------- pre-pass: per-expert [K][N] fp32 -> [N][K] bf16 transpose ----------------
__global__ __launch_bounds__(256)
void transpose_k(const float* __restrict__ src, unsigned short* __restrict__ dst,
                 int K, int N) {
    size_t eo = (size_t)blockIdx.z * K * N;
    const float* S = src + eo;
    unsigned short* T = dst + eo;
    int n0 = blockIdx.x * 32, k0 = blockIdx.y * 32;

    __shared__ unsigned short tile[32][33];
    int tid = threadIdx.x;
    int r = tid >> 3, c4 = (tid & 7) << 2;

    float4 v = *(const float4*)(S + (size_t)(k0 + r) * N + n0 + c4);
    tile[r][c4 + 0] = to_bf(v.x); tile[r][c4 + 1] = to_bf(v.y);
    tile[r][c4 + 2] = to_bf(v.z); tile[r][c4 + 3] = to_bf(v.w);
    __syncthreads();

    union { unsigned short s[4]; uint2 q; } pk;
    pk.s[0] = tile[c4 + 0][r]; pk.s[1] = tile[c4 + 1][r];
    pk.s[2] = tile[c4 + 2][r]; pk.s[3] = tile[c4 + 3][r];
    *(uint2*)(T + (size_t)(n0 + r) * K + k0 + c4) = pk.q;
}

// ---------------- kernel: gating (one wave32 per token) ----------------
__global__ __launch_bounds__(256)
void gate_k(const float* __restrict__ X, const float* __restrict__ gw,
            const float* __restrict__ gb, int* __restrict__ topk_id,
            float* __restrict__ topk_w, int* __restrict__ cnt) {
    int wid  = threadIdx.x >> 5;
    int lane = threadIdx.x & 31;
    int tok  = blockIdx.x * 8 + wid;

    float acc[8];
#pragma unroll
    for (int e = 0; e < 8; ++e) acc[e] = 0.f;

    const float* xr = X + (size_t)tok * DDIM;
    for (int d = lane; d < DDIM; d += 32) {
        float xv = xr[d];
        float4 g0 = *(const float4*)(gw + d * 8);
        float4 g1 = *(const float4*)(gw + d * 8 + 4);
        acc[0] += xv * g0.x; acc[1] += xv * g0.y;
        acc[2] += xv * g0.z; acc[3] += xv * g0.w;
        acc[4] += xv * g1.x; acc[5] += xv * g1.y;
        acc[6] += xv * g1.z; acc[7] += xv * g1.w;
    }
#pragma unroll
    for (int e = 0; e < 8; ++e) {
#pragma unroll
        for (int off = 16; off > 0; off >>= 1)
            acc[e] += __shfl_xor(acc[e], off);
    }
    if (lane == 0) {
        float l[8];
#pragma unroll
        for (int e = 0; e < 8; ++e) l[e] = acc[e] + gb[e];
        int i0 = 0;
#pragma unroll
        for (int e = 1; e < 8; ++e) if (l[e] > l[i0]) i0 = e;
        int i1 = (i0 == 0) ? 1 : 0;
#pragma unroll
        for (int e = 0; e < 8; ++e) if (e != i0 && l[e] > l[i1]) i1 = e;
        float e1 = expf(l[i1] - l[i0]);
        float inv = 1.f / (1.f + e1);
        topk_id[tok * 2 + 0] = i0;  topk_w[tok * 2 + 0] = inv;
        topk_id[tok * 2 + 1] = i1;  topk_w[tok * 2 + 1] = e1 * inv;
        atomicAdd(&cnt[i0], 1);
        atomicAdd(&cnt[i1], 1);
    }
}

// ---------------- tiny prefix + scatter ----------------
__global__ void prefix_k(const int* __restrict__ cnt, int* __restrict__ basep,
                         int* __restrict__ cursor) {
    if (threadIdx.x == 0) {
        int s = 0;
        for (int e = 0; e < NEXP; ++e) { basep[e] = s; cursor[e] = s; s += cnt[e]; }
    }
}

__global__ __launch_bounds__(256)
void scatter_k(const int* __restrict__ topk_id, const float* __restrict__ topk_w,
               int* __restrict__ cursor, int* __restrict__ bt, float* __restrict__ bw) {
    int n = blockIdx.x * blockDim.x + threadIdx.x;
    if (n >= NTOK) return;
#pragma unroll
    for (int k = 0; k < 2; ++k) {
        int e = topk_id[n * 2 + k];
        int s = atomicAdd(&cursor[e], 1);
        bt[s] = n;
        bw[s] = topk_w[n * 2 + k];
    }
}

// ---------------- GEMM tile geometry ----------------
// block tile 64(M) x 128(N), K-step 32; 8 waves, each wave owns 32x32 (=2x2 WMMA subtiles)
#define LDA 40                  // padded LDS row stride in bf16 elems (80B)
#define LDB 40
#define BUFB 15360u             // per double-buffer bytes: sB 128*80 + sA 64*80

// ---------------- kernel: h = gelu(gather(Xb) @ w1t^T + b1) ----------------
__global__ __launch_bounds__(256)
void gemm1_k(const unsigned short* __restrict__ Xb, const unsigned short* __restrict__ w1t,
             const float* __restrict__ b1, const int* __restrict__ bt,
             const int* __restrict__ cnt, const int* __restrict__ basep,
             unsigned short* __restrict__ h) {
    int e    = blockIdx.z;
    int rows = cnt[e];
    int r0   = blockIdx.x * 64;
    if (r0 >= rows) return;
    int rb = basep[e];
    int n0 = blockIdx.y * 128;

    extern __shared__ __align__(16) char smem[];   // dynamic LDS => base offset 0
    int* sTok = (int*)(smem + 2 * BUFB);

    int tid = threadIdx.x;
    if (tid < 64) {
        int r = r0 + tid;
        sTok[tid] = (r < rows) ? bt[rb + r] : -1;
    }
    __syncthreads();

    int lane = tid & 31, wid = tid >> 5;
    int wm = wid >> 2, wn = wid & 3;
    int lr = lane & 15, lh = lane >> 4;

    v8f z8 = {0.f, 0.f, 0.f, 0.f, 0.f, 0.f, 0.f, 0.f};
    v8f acc[2][2];
    acc[0][0] = z8; acc[0][1] = z8; acc[1][0] = z8; acc[1][1] = z8;

    int ar = tid >> 2, as = (tid & 3) << 3;
    int token = sTok[ar];
    const unsigned short* Bsrc = w1t + (size_t)e * HDIM * DDIM + (size_t)n0 * DDIM;

    auto stage = [&](int kt, int b) {
        unsigned ldsB = b * BUFB;
        unsigned ldsA = b * BUFB + 10240u;
        if (wid == 0)
            tdm_load_2d(ldsB, Bsrc + kt * 32, DDIM, HDIM, DDIM, 128);
        uint4 v = {0, 0, 0, 0};
        if (token >= 0)
            v = *(const uint4*)(Xb + (size_t)token * DDIM + kt * 32 + as);
        *(uint4*)((unsigned short*)(smem + ldsA) + ar * LDA + as) = v;
    };

    stage(0, 0);
    for (int kt = 0; kt < DDIM / 32; ++kt) {
        int b = kt & 1;
        if (wid == 0) __builtin_amdgcn_s_wait_tensorcnt(0);
        __syncthreads();

        const unsigned short* sB = (const unsigned short*)(smem + b * BUFB);
        const unsigned short* sA = (const unsigned short*)(smem + b * BUFB + 10240u);
        Frag fa[2], fb[2];
#pragma unroll
        for (int mi = 0; mi < 2; ++mi) {
            int row = wm * 32 + mi * 16 + lr;
            fa[mi].q[0] = *(const uint4*)&sA[row * LDA + lh * 8];
            fa[mi].q[1] = *(const uint4*)&sA[row * LDA + lh * 8 + 16];
        }
#pragma unroll
        for (int ni = 0; ni < 2; ++ni) {
            int col = wn * 32 + ni * 16 + lr;
            fb[ni].q[0] = *(const uint4*)&sB[col * LDB + lh * 8];
            fb[ni].q[1] = *(const uint4*)&sB[col * LDB + lh * 8 + 16];
        }
#pragma unroll
        for (int mi = 0; mi < 2; ++mi)
#pragma unroll
            for (int ni = 0; ni < 2; ++ni)
                acc[mi][ni] = __builtin_amdgcn_wmma_f32_16x16x32_bf16(
                    false, fa[mi].v, false, fb[ni].v, (short)0, acc[mi][ni],
                    false, false);

        if (kt + 1 < DDIM / 32) stage(kt + 1, b ^ 1);
    }

    // epilogue: bias + gelu -> h (bf16)
#pragma unroll
    for (int mi = 0; mi < 2; ++mi) {
#pragma unroll
        for (int ni = 0; ni < 2; ++ni) {
            int col = n0 + wn * 32 + ni * 16 + lr;
            float bias = b1[e * HDIM + col];
#pragma unroll
            for (int j = 0; j < 8; ++j) {
                int m = wm * 32 + mi * 16 + (lh << 3) + j;
                int r = r0 + m;
                if (r < rows) {
                    float x = acc[mi][ni][j] + bias;
                    h[(size_t)(rb + r) * HDIM + col] = to_bf(gelu_tanh(x));
                }
            }
        }
    }
}

// ---------------- kernel: out += w * (h @ w2t^T + b2) ----------------
__global__ __launch_bounds__(256)
void gemm2_k(const unsigned short* __restrict__ h, const unsigned short* __restrict__ w2t,
             const float* __restrict__ b2, const int* __restrict__ bt,
             const float* __restrict__ bw, const int* __restrict__ cnt,
             const int* __restrict__ basep, float* __restrict__ out) {
    int e    = blockIdx.z;
    int rows = cnt[e];
    int r0   = blockIdx.x * 64;
    if (r0 >= rows) return;
    int rb = basep[e];
    int n0 = blockIdx.y * 128;

    extern __shared__ __align__(16) char smem[];
    int*   sTok = (int*)(smem + 2 * BUFB);
    float* sW   = (float*)(smem + 2 * BUFB + 256);

    int tid = threadIdx.x;
    if (tid < 64) {
        int r = r0 + tid;
        sTok[tid] = (r < rows) ? bt[rb + r] : 0;
        sW[tid]   = (r < rows) ? bw[rb + r] : 0.f;
    }
    __syncthreads();

    int lane = tid & 31, wid = tid >> 5;
    int wm = wid >> 2, wn = wid & 3;
    int lr = lane & 15, lh = lane >> 4;

    v8f z8 = {0.f, 0.f, 0.f, 0.f, 0.f, 0.f, 0.f, 0.f};
    v8f acc[2][2];
    acc[0][0] = z8; acc[0][1] = z8; acc[1][0] = z8; acc[1][1] = z8;

    const unsigned short* Bsrc = w2t + (size_t)e * DDIM * HDIM + (size_t)n0 * HDIM;
    const unsigned short* Asrc = h + (size_t)(rb + r0) * HDIM;   // contiguous slots

    auto stage = [&](int kt, int b) {
        if (wid == 0) {
            tdm_load_2d(b * BUFB,          Bsrc + kt * 32, HDIM, DDIM,          HDIM, 128);
            tdm_load_2d(b * BUFB + 10240u, Asrc + kt * 32, HDIM, 2 * NTOK + 64, HDIM, 64);
        }
    };

    stage(0, 0);
    for (int kt = 0; kt < HDIM / 32; ++kt) {
        int b = kt & 1;
        if (wid == 0) __builtin_amdgcn_s_wait_tensorcnt(0);
        __syncthreads();

        const unsigned short* sB = (const unsigned short*)(smem + b * BUFB);
        const unsigned short* sA = (const unsigned short*)(smem + b * BUFB + 10240u);
        Frag fa[2], fb[2];
#pragma unroll
        for (int mi = 0; mi < 2; ++mi) {
            int row = wm * 32 + mi * 16 + lr;
            fa[mi].q[0] = *(const uint4*)&sA[row * LDA + lh * 8];
            fa[mi].q[1] = *(const uint4*)&sA[row * LDA + lh * 8 + 16];
        }
#pragma unroll
        for (int ni = 0; ni < 2; ++ni) {
            int col = wn * 32 + ni * 16 + lr;
            fb[ni].q[0] = *(const uint4*)&sB[col * LDB + lh * 8];
            fb[ni].q[1] = *(const uint4*)&sB[col * LDB + lh * 8 + 16];
        }
#pragma unroll
        for (int mi = 0; mi < 2; ++mi)
#pragma unroll
            for (int ni = 0; ni < 2; ++ni)
                acc[mi][ni] = __builtin_amdgcn_wmma_f32_16x16x32_bf16(
                    false, fa[mi].v, false, fb[ni].v, (short)0, acc[mi][ni],
                    false, false);

        if (kt + 1 < HDIM / 32) stage(kt + 1, b ^ 1);
    }

    // epilogue: bias, combine weight, commutative atomic scatter-add
#pragma unroll
    for (int mi = 0; mi < 2; ++mi) {
#pragma unroll
        for (int ni = 0; ni < 2; ++ni) {
            int col = n0 + wn * 32 + ni * 16 + lr;
            float bias = b2[e * DDIM + col];
#pragma unroll
            for (int j = 0; j < 8; ++j) {
                int m = wm * 32 + mi * 16 + (lh << 3) + j;
                int r = r0 + m;
                if (r < rows) {
                    float y = acc[mi][ni][j] + bias;
                    atomicAdd(&out[(size_t)sTok[m] * DDIM + col], sW[m] * y);
                }
            }
        }
    }
}

// ---------------- launch ----------------
extern "C" void kernel_launch(void* const* d_in, const int* in_sizes, int n_in,
                              void* d_out, int out_size, void* d_ws, size_t ws_size,
                              hipStream_t stream) {
    (void)in_sizes; (void)n_in; (void)out_size; (void)ws_size;
    const float* X  = (const float*)d_in[0];
    const float* gw = (const float*)d_in[1];
    const float* gb = (const float*)d_in[2];
    const float* w1 = (const float*)d_in[3];
    const float* b1 = (const float*)d_in[4];
    const float* w2 = (const float*)d_in[5];
    const float* b2 = (const float*)d_in[6];
    float* out = (float*)d_out;

    // workspace layout (~273 MB assumed available)
    char* ws = (char*)d_ws;
    int*   cnt     = (int*)ws;
    int*   basep   = cnt + 8;
    int*   cursor  = cnt + 16;
    int*   topk_id = cnt + 32;
    float* topk_w  = (float*)(topk_id + 2 * NTOK);
    int*   bt      = (int*)(topk_w + 2 * NTOK);
    float* bw      = (float*)(bt + 2 * NTOK);
    size_t off = 1u << 19;                                   // 512 KB
    unsigned short* Xb  = (unsigned short*)(ws + off);       off += (size_t)NTOK * DDIM * 2;
    unsigned short* w1t = (unsigned short*)(ws + off);       off += (size_t)NEXP * DDIM * HDIM * 2;
    unsigned short* w2t = (unsigned short*)(ws + off);       off += (size_t)NEXP * DDIM * HDIM * 2;
    unsigned short* hb  = (unsigned short*)(ws + off);       // [2N+64][H] bf16 (+slack rows)

    (void)hipMemsetAsync(cnt, 0, 128, stream);
    (void)hipMemsetAsync(out, 0, (size_t)NTOK * DDIM * sizeof(float), stream);

    convx_k    <<<(NTOK * DDIM / 8) / 256, 256, 0, stream>>>(X, Xb);
    transpose_k<<<dim3(HDIM / 32, DDIM / 32, NEXP), 256, 0, stream>>>(w1, w1t, DDIM, HDIM);
    transpose_k<<<dim3(DDIM / 32, HDIM / 32, NEXP), 256, 0, stream>>>(w2, w2t, HDIM, DDIM);

    gate_k   <<<NTOK / 8,   256, 0, stream>>>(X, gw, gb, topk_id, topk_w, cnt);
    prefix_k <<<1,          32,  0, stream>>>(cnt, basep, cursor);
    scatter_k<<<NTOK / 256, 256, 0, stream>>>(topk_id, topk_w, cursor, bt, bw);

    unsigned smem = 2 * BUFB + 512;   // double buffers + sTok/sW
    gemm1_k<<<dim3(NTOK / 64, HDIM / 128, NEXP), 256, smem, stream>>>(
        Xb, w1t, b1, bt, cnt, basep, hb);
    gemm2_k<<<dim3(NTOK / 64, DDIM / 128, NEXP), 256, smem, stream>>>(
        hb, w2t, b2, bt, bw, cnt, basep, out);
}
